// MultiHeadAttention_82952998355870
// MI455X (gfx1250) — matrix operand order repaired and checked by reading, not compiled, not run on hardware
//
#include <hip/hip_runtime.h>

// ---------------------------------------------------------------------------
// MI455X (gfx1250) multi-head attention, f16 WMMA pipeline with f32 accum.
// Stages: cvt/transpose -> QKV WMMA GEMMs -> flash-attention (WMMA) -> out GEMM
// GEMM uses a 2-deep register pipeline so fragment loads overlap the WMMAs.
// ---------------------------------------------------------------------------

typedef __attribute__((ext_vector_type(16))) _Float16 hv16;
typedef __attribute__((ext_vector_type(8)))  _Float16 hv8;
typedef __attribute__((ext_vector_type(8)))  float    fv8;

#define S_LEN  2048
#define DEMB   1024
#define NHEAD  16
#define HDIM   64
#define BATCH  2
#define MROWS  (BATCH * S_LEN)   // 4096

// --------------------------- WMMA helpers ----------------------------------

__device__ __forceinline__ fv8 wmma16(hv16 a, hv16 b, fv8 c) {
  // v_wmma_f32_16x16x32_f16 (neg_a, A, neg_b, B, c_mod, C, reuse_a, reuse_b)
  return __builtin_amdgcn_wmma_f32_16x16x32_f16(false, a, false, b, (short)0, c,
                                                false, false);
}

// A-fragment 16x32 f16, source row-major [16 rows x >=32 cols], ld in elements.
// Lane L: m = L%16, half = L/16.  elem e -> K = 16*half + e (contiguous per lane).
__device__ __forceinline__ hv16 frag_a_load(const _Float16* p, int ld) {
  const int lane = threadIdx.x & 31;
  const int m = lane & 15, h = lane >> 4;
  const _Float16* r = p + (size_t)m * ld;
  hv8 lo = *(const hv8*)(r + h * 16);
  hv8 hi = *(const hv8*)(r + h * 16 + 8);
  hv16 a;
#pragma unroll
  for (int i = 0; i < 8; ++i) { a[i] = lo[i]; a[8 + i] = hi[i]; }
  return a;
}

// B-fragment 32x16 f16 where memory holds B transposed: [16 N-rows x >=32 K-cols].
// Lane L: n = L%16, half = L/16. elem e -> K = 16*half + e (contiguous per lane).
__device__ __forceinline__ hv16 frag_b_load(const _Float16* p, int ld) {
  const int lane = threadIdx.x & 31;
  const int n = lane & 15, h = lane >> 4;
  const _Float16* r = p + (size_t)n * ld + h * 16;
  hv8 lo = *(const hv8*)(r);
  hv8 hi = *(const hv8*)(r + 8);
  hv16 b;
#pragma unroll
  for (int i = 0; i < 8; ++i) { b[i] = lo[i]; b[8 + i] = hi[i]; }
  return b;
}

// --------------------------- conversion kernels ----------------------------

__global__ void __launch_bounds__(256) cvt_f32_f16_kernel(const float* __restrict__ s,
                                                          _Float16* __restrict__ d,
                                                          int nElem) {
  int i = blockIdx.x * 256 + threadIdx.x;
  if (i < nElem) d[i] = (_Float16)s[i];
}

// dst[n*K + k] = (f16) src[k*N + n]   (weight -> transposed f16)
__global__ void __launch_bounds__(256) transpose_cvt_kernel(const float* __restrict__ s,
                                                            _Float16* __restrict__ d,
                                                            int K, int N) {
  int i = blockIdx.x * 256 + threadIdx.x;
  if (i < K * N) {
    int n = i / K, k = i % K;
    d[i] = (_Float16)s[(size_t)k * N + n];
  }
}

// --------------------------- WMMA GEMM -------------------------------------
// Y[M,N] = X[M,K] @ W[K,N] + bias ; W given transposed (Wt[N,K], f16).
// WG: 256 threads = 8 waves in 2(M) x 4(N) grid; WG tile 64M x 128N;
// wave tile 32M x 32N (2x2 WMMA tiles, each fragment reused twice).
// 2-deep software pipeline: fragments for k+32 are in flight during the
// WMMAs of step k, so s_wait_loadcnt overlaps the matrix pipe.
// mode 0: f32 row-major [M,N] ; mode 1: f16 [b,h,s,hd] ; mode 2: f16 [b,h,hd,s]
__global__ void __launch_bounds__(256) gemm_wmma_kernel(
    const _Float16* __restrict__ X, const _Float16* __restrict__ Wt,
    const float* __restrict__ bias, void* __restrict__ outp,
    int Kd, int N, int mode) {
  const int wave = threadIdx.x >> 5;
  const int lane = threadIdx.x & 31;
  const int wm = wave >> 2;            // 0..1
  const int wn = wave & 3;             // 0..3
  const int m0 = blockIdx.x * 64 + wm * 32;
  const int n0 = blockIdx.y * 128 + wn * 32;

  const _Float16* xr0 = X + (size_t)m0 * Kd;
  const _Float16* xr1 = xr0 + (size_t)16 * Kd;
  const _Float16* wr0 = Wt + (size_t)n0 * Kd;
  const _Float16* wr1 = wr0 + (size_t)16 * Kd;

  fv8 acc00 = {}, acc01 = {}, acc10 = {}, acc11 = {};

  hv16 aN0 = frag_a_load(xr0, Kd);
  hv16 aN1 = frag_a_load(xr1, Kd);
  hv16 bN0 = frag_b_load(wr0, Kd);
  hv16 bN1 = frag_b_load(wr1, Kd);

  for (int k = 0; k < Kd; k += 32) {
    hv16 a0 = aN0, a1 = aN1, b0 = bN0, b1 = bN1;
    const int kn = k + 32;
    if (kn < Kd) {                      // prefetch next step's fragments
      aN0 = frag_a_load(xr0 + kn, Kd);
      aN1 = frag_a_load(xr1 + kn, Kd);
      bN0 = frag_b_load(wr0 + kn, Kd);
      bN1 = frag_b_load(wr1 + kn, Kd);
    }
    acc00 = wmma16(a0, b0, acc00);
    acc01 = wmma16(a0, b1, acc01);
    acc10 = wmma16(a1, b0, acc10);
    acc11 = wmma16(a1, b1, acc11);
  }

  const int n = lane & 15, h = lane >> 4;
  fv8 accs[2][2] = {{acc00, acc01}, {acc10, acc11}};
#pragma unroll
  for (int nt = 0; nt < 2; ++nt) {
    const int col = n0 + nt * 16 + n;
    const float bval = bias[col];
#pragma unroll
    for (int mt = 0; mt < 2; ++mt) {
#pragma unroll
      for (int r = 0; r < 8; ++r) {
        const int row = m0 + mt * 16 + r + 8 * h;
        const float v = accs[mt][nt][r] + bval;
        if (mode == 0) {
          ((float*)outp)[(size_t)row * N + col] = v;
        } else {
          const int b  = row >> 11, s = row & (S_LEN - 1);
          const int hh = col >> 6,  hd = col & (HDIM - 1);
          _Float16* o = (_Float16*)outp;
          if (mode == 1)
            o[(((size_t)(b * NHEAD + hh)) * S_LEN + s) * HDIM + hd] = (_Float16)v;
          else
            o[(((size_t)(b * NHEAD + hh)) * HDIM + hd) * S_LEN + s] = (_Float16)v;
        }
      }
    }
  }
}

// --------------------------- flash attention -------------------------------
// Q,K: f16 [b,h,s,hd] ; Vt: f16 [b,h,hd,s] ; ctx out: f16 [b,s,h*hd]
// One wave handles 16 query rows of one (b,h); streams keys in blocks of 32.
// V fragments are loaded before the softmax so they overlap the VALU work;
// next block's K/V lines are prefetched (global_prefetch_b8).
__global__ void __launch_bounds__(256) attn_wmma_kernel(
    const _Float16* __restrict__ Q, const _Float16* __restrict__ K,
    const _Float16* __restrict__ Vt, _Float16* __restrict__ ctx) {
  __shared__ __align__(16) _Float16 pbuf[8 * 16 * 32];  // 1KB P-slot per wave

  const int wave = threadIdx.x >> 5;
  const int lane = threadIdx.x & 31;
  const int task = blockIdx.x * 8 + wave;          // 0 .. B*H*(S/16)-1
  const int qt = task & (S_LEN / 16 - 1);          // query tile in head
  const int bh = task >> 7;                        // 0..31 = b*16 + h

  const _Float16* qp = Q  + (size_t)bh * S_LEN * HDIM + (size_t)qt * 16 * HDIM;
  const _Float16* kp = K  + (size_t)bh * S_LEN * HDIM;
  const _Float16* vp = Vt + (size_t)bh * HDIM * S_LEN;
  _Float16* ps = pbuf + wave * (16 * 32);
  const int n = lane & 15, h = lane >> 4;

  // Q A-fragments for the two 32-wide K-dim steps of HD=64 (kept in VGPRs)
  const hv16 aq0 = frag_a_load(qp, HDIM);
  const hv16 aq1 = frag_a_load(qp + 32, HDIM);

  fv8 o[4] = {{}, {}, {}, {}};
  float mi[8], li[8];
#pragma unroll
  for (int r = 0; r < 8; ++r) { mi[r] = -1e30f; li[r] = 0.f; }
  const float scale = 0.125f;  // 1/sqrt(64)

  for (int j = 0; j < S_LEN; j += 32) {
    // ---- prefetch next key block (K: 32 rows, one per lane; V: 2 hd-rows) --
    if (j + 32 < S_LEN) {
      __builtin_prefetch(kp + (size_t)(j + 32 + lane) * HDIM, 0, 0);
      __builtin_prefetch(vp + (size_t)lane * S_LEN + j + 32, 0, 0);
      __builtin_prefetch(vp + (size_t)(lane + 32) * S_LEN + j + 32, 0, 0);
    }

    // ---- S = Q @ K^T for two 16-key tiles ----
    fv8 s0 = {}, s1 = {};
    {
      hv16 b0 = frag_b_load(kp + (size_t)j * HDIM, HDIM);
      hv16 b1 = frag_b_load(kp + (size_t)j * HDIM + 32, HDIM);
      hv16 b2 = frag_b_load(kp + (size_t)(j + 16) * HDIM, HDIM);
      hv16 b3 = frag_b_load(kp + (size_t)(j + 16) * HDIM + 32, HDIM);
      s0 = wmma16(aq0, b0, s0);
      s0 = wmma16(aq1, b1, s0);
      s1 = wmma16(aq0, b2, s1);
      s1 = wmma16(aq1, b3, s1);
    }

    // ---- V fragments now: loads overlap the softmax VALU below ----
    hv16 bv[4];
#pragma unroll
    for (int t = 0; t < 4; ++t)
      bv[t] = frag_b_load(vp + (size_t)(t * 16) * S_LEN + j, S_LEN);

    // ---- online softmax (rows r + 8*half; reduce across the 16-lane half) --
    float mnew[8];
#pragma unroll
    for (int r = 0; r < 8; ++r) {
      s0[r] *= scale; s1[r] *= scale;
      mnew[r] = fmaxf(s0[r], s1[r]);
    }
#pragma unroll
    for (int d = 1; d < 16; d <<= 1)
#pragma unroll
      for (int r = 0; r < 8; ++r)
        mnew[r] = fmaxf(mnew[r], __shfl_xor(mnew[r], d, 32));
    float alpha[8], rsum[8];
#pragma unroll
    for (int r = 0; r < 8; ++r) {
      const float mn = fmaxf(mi[r], mnew[r]);
      alpha[r] = __expf(mi[r] - mn);
      mi[r] = mn;
      const float p0 = __expf(s0[r] - mn);
      const float p1 = __expf(s1[r] - mn);
      s0[r] = p0; s1[r] = p1;
      rsum[r] = p0 + p1;
    }
#pragma unroll
    for (int d = 1; d < 16; d <<= 1)
#pragma unroll
      for (int r = 0; r < 8; ++r)
        rsum[r] += __shfl_xor(rsum[r], d, 32);
#pragma unroll
    for (int r = 0; r < 8; ++r) li[r] = li[r] * alpha[r] + rsum[r];
#pragma unroll
    for (int t = 0; t < 4; ++t)
#pragma unroll
      for (int r = 0; r < 8; ++r) o[t][r] *= alpha[r];

    // ---- repack P (C-layout) -> A-layout through wave-private LDS slot ----
#pragma unroll
    for (int r = 0; r < 8; ++r) {
      const int row = r + 8 * h;
      ps[row * 32 + n]      = (_Float16)s0[r];
      ps[row * 32 + 16 + n] = (_Float16)s1[r];
    }
    asm volatile("s_wait_dscnt 0x0" ::: "memory");
    hv16 ap = frag_a_load(ps, 32);

    // ---- O += P @ V ----
#pragma unroll
    for (int t = 0; t < 4; ++t)
      o[t] = wmma16(ap, bv[t], o[t]);
  }

  // ---- normalize and write ctx [b, s, h*64 + hd] (f16) ----
  const int b = bh >> 4, hh = bh & 15;
#pragma unroll
  for (int r = 0; r < 8; ++r) {
    const float inv = 1.0f / li[r];
    const int s = qt * 16 + r + 8 * h;
    const size_t base = ((size_t)(b * S_LEN + s)) * DEMB + hh * HDIM;
#pragma unroll
    for (int t = 0; t < 4; ++t)
      ctx[base + t * 16 + n] = (_Float16)(o[t][r] * inv);
  }
}

// --------------------------- launcher --------------------------------------

extern "C" void kernel_launch(void* const* d_in, const int* in_sizes, int n_in,
                              void* d_out, int out_size, void* d_ws, size_t ws_size,
                              hipStream_t stream) {
  (void)in_sizes; (void)n_in; (void)out_size; (void)ws_size;
  const float* inp = (const float*)d_in[0];
  const float* Wq  = (const float*)d_in[1];
  const float* bq  = (const float*)d_in[2];
  const float* Wk  = (const float*)d_in[3];
  const float* bk  = (const float*)d_in[4];
  const float* Wv  = (const float*)d_in[5];
  const float* bv  = (const float*)d_in[6];
  const float* Wo  = (const float*)d_in[7];
  const float* bo  = (const float*)d_in[8];

  char* ws = (char*)d_ws;
  const size_t SZ_X = (size_t)MROWS * DEMB * sizeof(_Float16);   // 8 MB
  const size_t SZ_W = (size_t)DEMB * DEMB * sizeof(_Float16);    // 2 MB
  _Float16* Xh   = (_Float16*)(ws);
  _Float16* Wqt  = (_Float16*)(ws + SZ_X);
  _Float16* Wkt  = (_Float16*)(ws + SZ_X + SZ_W);
  _Float16* Wvt  = (_Float16*)(ws + SZ_X + 2 * SZ_W);
  _Float16* Wot  = (_Float16*)(ws + SZ_X + 3 * SZ_W);
  _Float16* Qh   = (_Float16*)(ws + SZ_X + 4 * SZ_W);
  _Float16* Kh   = (_Float16*)(ws + 2 * SZ_X + 4 * SZ_W);
  _Float16* Vth  = (_Float16*)(ws + 3 * SZ_X + 4 * SZ_W);
  _Float16* Ctxh = (_Float16*)(ws + 4 * SZ_X + 4 * SZ_W);        // total 48 MB

  // Stage 1: precision conversion (+ weight transpose)
  cvt_f32_f16_kernel<<<(MROWS * DEMB) / 256, 256, 0, stream>>>(inp, Xh, MROWS * DEMB);
  transpose_cvt_kernel<<<(DEMB * DEMB) / 256, 256, 0, stream>>>(Wq, Wqt, DEMB, DEMB);
  transpose_cvt_kernel<<<(DEMB * DEMB) / 256, 256, 0, stream>>>(Wk, Wkt, DEMB, DEMB);
  transpose_cvt_kernel<<<(DEMB * DEMB) / 256, 256, 0, stream>>>(Wv, Wvt, DEMB, DEMB);
  transpose_cvt_kernel<<<(DEMB * DEMB) / 256, 256, 0, stream>>>(Wo, Wot, DEMB, DEMB);

  // Stage 2: QKV projections (WMMA), epilogue scatters into attention layouts
  dim3 gg(MROWS / 64, DEMB / 128);
  gemm_wmma_kernel<<<gg, 256, 0, stream>>>(Xh, Wqt, bq, Qh,  DEMB, DEMB, 1);
  gemm_wmma_kernel<<<gg, 256, 0, stream>>>(Xh, Wkt, bk, Kh,  DEMB, DEMB, 1);
  gemm_wmma_kernel<<<gg, 256, 0, stream>>>(Xh, Wvt, bv, Vth, DEMB, DEMB, 2);

  // Stage 3: flash attention (WMMA QK^T + PV, online softmax)
  attn_wmma_kernel<<<(BATCH * NHEAD * (S_LEN / 16)) / 8, 256, 0, stream>>>(Qh, Kh, Vth, Ctxh);

  // Stage 4: output projection, f32 + bias straight to d_out
  gemm_wmma_kernel<<<gg, 256, 0, stream>>>(Ctxh, Wot, bo, d_out, DEMB, DEMB, 0);
}
